// Critic_19164144075376
// MI455X (gfx1250) — matrix operand order repaired and checked by reading, not compiled, not run on hardware
//
#include <hip/hip_runtime.h>
#include <hip/hip_bf16.h>

typedef __attribute__((ext_vector_type(16))) __bf16 v16bf;
typedef __attribute__((ext_vector_type(8)))  __bf16 v8bf;
typedef __attribute__((ext_vector_type(8)))  float  v8f;

#define N_NODES 100000
#define N_EDGES 1600000
#define G_BATCH 64
#define F_DIM   128
#define SV_DIM  64
#define AD_DIM  32
#define FC_DIM  256
#define POOL_CHUNK 512

static __device__ __forceinline__ void atomic_add_f32(float* p, float v) {
    unsafeAtomicAdd(p, v);   // lowers to global_atomic_add_f32 on CDNA
}

// ---------------- utility kernels ----------------

__global__ void fill_f32(float* __restrict__ p, float v, int n) {
    int i = blockIdx.x * blockDim.x + threadIdx.x;
    if (i < n) p[i] = v;
}

__global__ void deg_accum(const int* __restrict__ dst, float* __restrict__ deg, int e) {
    int i = blockIdx.x * blockDim.x + threadIdx.x;
    if (i < e) atomic_add_f32(&deg[dst[i]], 1.0f);
}

__global__ void deg_rsqrt(float* __restrict__ deg, int n) {
    int i = blockIdx.x * blockDim.x + threadIdx.x;
    if (i < n) deg[i] = rsqrtf(deg[i]);
}

__global__ void cvt_f32_bf16(const float* __restrict__ in, __bf16* __restrict__ out, int n) {
    int i = blockIdx.x * blockDim.x + threadIdx.x;
    if (i < n) out[i] = (__bf16)in[i];
}

// Fused f32->bf16 convert + pack of W[128x128] into WMMA B-fragment order:
// idx = ((wt*4 + kb)*32 + lane)*16 + e  ->  B element (k = kb*32 + (lane>>4)*16 + e,
//                                            col = wt*16 + (lane&15))
__global__ void pack_w_bf16(const float* __restrict__ W, __bf16* __restrict__ Bp) {
    int idx = blockIdx.x * blockDim.x + threadIdx.x;
    if (idx >= 8 * 4 * 32 * 16) return;
    int e  = idx & 15;
    int l  = (idx >> 4) & 31;
    int kb = (idx >> 9) & 3;
    int wt = idx >> 11;
    int k   = kb * 32 + (l >> 4) * 16 + e;
    int col = wt * 16 + (l & 15);
    Bp[idx] = (__bf16)W[k * F_DIM + col];
}

// ---------------- WMMA GEMM: C[M x 128] = A[M x 128] (bf16) * W (packed bf16) ----------------
// 256 threads = 8 waves; wave w owns columns [16w,16w+16). A panel staged in LDS once per
// block (128 rows); B fragments come pre-packed -> pure vector loads.

__global__ __launch_bounds__(256) void gemm_bf16_wmma(
    const __bf16* __restrict__ A, const __bf16* __restrict__ Bp,
    float* __restrict__ C, int M)
{
    __shared__ __bf16 As[128 * F_DIM];          // 32 KB
    const int tid  = threadIdx.x;
    const int lane = tid & 31;
    const int wave = tid >> 5;                  // 0..7 -> column tile
    const int m    = lane & 15;
    const int half = lane >> 4;
    const int col0 = wave * 16;
    const int rowBase = blockIdx.x * 128;

    // Packed B fragments: one 32-byte vector per (wave, chunk, lane).
    v16bf bfrag[4];
#pragma unroll
    for (int kb = 0; kb < 4; ++kb)
        bfrag[kb] = *(const v16bf*)(Bp + (size_t)((wave * 4 + kb) * 32 + lane) * 16);

    // Stage A panel into LDS: 8 iters x 256 threads x 16B = 32 KB.
#pragma unroll
    for (int it = 0; it < 8; ++it) {
        int elt = (it * 256 + tid) * 8;         // bf16 element offset (8 elems = 16B)
        int row = rowBase + (elt >> 7);
        v8bf v = {};
        if (row < M) v = *(const v8bf*)(A + (size_t)rowBase * F_DIM + elt);
        *(v8bf*)(As + elt) = v;
    }
    __syncthreads();

#pragma unroll
    for (int t = 0; t < 8; ++t) {
        int row0 = rowBase + t * 16;
        if (row0 >= M) break;                   // uniform; EXEC stays all-ones
        const __bf16* Arow = As + (size_t)(t * 16 + m) * F_DIM;
        v8f c = {};
#pragma unroll
        for (int kb = 0; kb < 4; ++kb) {
            v8bf lo = *(const v8bf*)(Arow + kb * 32 + half * 16);
            v8bf hi = *(const v8bf*)(Arow + kb * 32 + half * 16 + 8);
            v16bf a;
#pragma unroll
            for (int e = 0; e < 8; ++e) { a[e] = lo[e]; a[8 + e] = hi[e]; }
            c = __builtin_amdgcn_wmma_f32_16x16x32_bf16(
                    false, a, false, bfrag[kb], (short)0, c, false, false);
        }
        float* Crow = C + (size_t)row0 * F_DIM + col0 + m;
#pragma unroll
        for (int r = 0; r < 8; ++r)
            Crow[(size_t)(half * 8 + r) * F_DIM] = c[r];
    }
}

// ---------------- edge scatter: agg[dst] += h[src] * dis[src]*dis[dst] ----------------
// 32 lanes per edge, float4 per lane: vectorized gather, 4x fewer index loads.

__global__ void edge_agg(const float4* __restrict__ h4, const int* __restrict__ src,
                         const int* __restrict__ dst, const float* __restrict__ dis,
                         float* __restrict__ agg)
{
    const long long total  = (long long)N_EDGES * 32;
    const long long stride = (long long)gridDim.x * blockDim.x;
    for (long long idx = (long long)blockIdx.x * blockDim.x + threadIdx.x;
         idx < total; idx += stride) {
        int e = (int)(idx >> 5);
        int q = (int)(idx & 31);
        int s = src[e], d = dst[e];
        float nrm = dis[s] * dis[d];
        float4 v = h4[(size_t)s * 32 + q];
        float* ap = agg + (size_t)d * F_DIM + q * 4;
        atomic_add_f32(ap + 0, v.x * nrm);
        atomic_add_f32(ap + 1, v.y * nrm);
        atomic_add_f32(ap + 2, v.z * nrm);
        atomic_add_f32(ap + 3, v.w * nrm);
    }
}

// self-loop + bias + ReLU; keep f32 result in agg, emit bf16 copy for next layer's GEMM
__global__ void gcn_post(float* __restrict__ agg, const float* __restrict__ h,
                         const float* __restrict__ dis, const float* __restrict__ bias,
                         __bf16* __restrict__ featbf)
{
    int idx = blockIdx.x * blockDim.x + threadIdx.x;
    if (idx >= N_NODES * F_DIM) return;
    int nd = idx >> 7, f = idx & 127;
    float v = agg[idx] + h[idx] * dis[nd] * dis[nd] + bias[f];
    v = fmaxf(v, 0.0f);
    agg[idx] = v;
    featbf[idx] = (__bf16)v;
}

// ---------------- pooling (batch is sorted -> chunked running sum) ----------------

__global__ __launch_bounds__(F_DIM) void pool_sum(const float* __restrict__ h,
                                                  const int* __restrict__ batch,
                                                  float* __restrict__ pooled)
{
    int f  = threadIdx.x;
    int n0 = blockIdx.x * POOL_CHUNK;
    if (n0 >= N_NODES) return;
    int n1 = n0 + POOL_CHUNK;
    if (n1 > N_NODES) n1 = N_NODES;
    int cur = batch[n0];
    float acc = 0.0f;
    for (int n = n0; n < n1; ++n) {
        int g = batch[n];
        if (g != cur) {
            atomic_add_f32(&pooled[cur * F_DIM + f], acc);
            acc = 0.0f;
            cur = g;
        }
        acc += h[(size_t)n * F_DIM + f];
    }
    atomic_add_f32(&pooled[cur * F_DIM + f], acc);
}

__global__ void pool_cnt(const int* __restrict__ batch, float* __restrict__ cnt) {
    int i = blockIdx.x * blockDim.x + threadIdx.x;
    if (i < N_NODES) atomic_add_f32(&cnt[batch[i]], 1.0f);
}

__global__ void pool_div(float* __restrict__ pooled, const float* __restrict__ cnt) {
    int idx = blockIdx.x * blockDim.x + threadIdx.x;
    if (idx >= G_BATCH * F_DIM) return;
    pooled[idx] /= fmaxf(cnt[idx >> 7], 1.0f);
}

// ---------------- MLP head (G=64 rows, tiny) ----------------

__global__ __launch_bounds__(FC_DIM) void mlp1(
    const float* __restrict__ pooled, const float* __restrict__ sv,
    const float* __restrict__ act, const float* __restrict__ W,
    const float* __restrict__ b, float* __restrict__ z)
{
    int g = blockIdx.x, j = threadIdx.x;
    float acc = b[j];
    for (int k = 0; k < F_DIM; ++k)  acc += pooled[g * F_DIM + k] * W[k * FC_DIM + j];
    for (int k = 0; k < SV_DIM; ++k) acc += sv[g * SV_DIM + k] * W[(F_DIM + k) * FC_DIM + j];
    for (int k = 0; k < AD_DIM; ++k) acc += act[g * AD_DIM + k] * W[(F_DIM + SV_DIM + k) * FC_DIM + j];
    z[g * FC_DIM + j] = fmaxf(acc, 0.0f);
}

__global__ __launch_bounds__(FC_DIM) void mlp2(
    const float* __restrict__ zin, const float* __restrict__ W,
    const float* __restrict__ b, float* __restrict__ z)
{
    int g = blockIdx.x, j = threadIdx.x;
    float acc = b[j];
    for (int k = 0; k < FC_DIM; ++k) acc += zin[g * FC_DIM + k] * W[k * FC_DIM + j];
    z[g * FC_DIM + j] = fmaxf(acc, 0.0f);
}

__global__ void mlp_out(const float* __restrict__ zin, const float* __restrict__ Wo,
                        const float* __restrict__ bo, float* __restrict__ out)
{
    int g = threadIdx.x;
    if (g >= G_BATCH) return;
    float acc = bo[0];
    for (int k = 0; k < FC_DIM; ++k) acc += zin[g * FC_DIM + k] * Wo[k];
    out[g] = acc;
}

// ---------------- host launcher ----------------

extern "C" void kernel_launch(void* const* d_in, const int* in_sizes, int n_in,
                              void* d_out, int out_size, void* d_ws, size_t ws_size,
                              hipStream_t stream)
{
    (void)in_sizes; (void)n_in; (void)out_size; (void)ws_size;

    const float* x     = (const float*)d_in[0];
    const int*   ei    = (const int*)d_in[1];     // [2, E] flat: src then dst
    const int*   batch = (const int*)d_in[2];
    const float* sv    = (const float*)d_in[3];
    const float* act   = (const float*)d_in[4];
    const float* W1    = (const float*)d_in[5];
    const float* b1    = (const float*)d_in[6];
    const float* W2    = (const float*)d_in[7];
    const float* b2    = (const float*)d_in[8];
    const float* Wf1   = (const float*)d_in[9];
    const float* bf1   = (const float*)d_in[10];
    const float* Wf2   = (const float*)d_in[11];
    const float* bf2   = (const float*)d_in[12];
    const float* Wo    = (const float*)d_in[13];
    const float* bo    = (const float*)d_in[14];
    float* out = (float*)d_out;

    const int* src = ei;
    const int* dst = ei + N_EDGES;

    // workspace carve-up (256B aligned)
    char* ws = (char*)d_ws;
    size_t off = 0;
    auto carve = [&](size_t bytes) {
        void* p = ws + off;
        off = (off + bytes + 255) & ~(size_t)255;
        return p;
    };
    float*  deg    = (float*) carve((size_t)N_NODES * 4);                 // deg -> deg_inv_sqrt
    __bf16* featbf = (__bf16*)carve((size_t)N_NODES * F_DIM * 2);         // bf16 layer input
    float*  hbuf   = (float*) carve((size_t)N_NODES * F_DIM * 4);         // GEMM output
    float*  aggbuf = (float*) carve((size_t)N_NODES * F_DIM * 4);         // aggregation / layer out
    __bf16* wpk    = (__bf16*)carve((size_t)8 * 4 * 32 * 16 * 2);         // packed bf16 weights
    float*  pooled = (float*) carve((size_t)G_BATCH * F_DIM * 4);
    float*  cnt    = (float*) carve((size_t)G_BATCH * 4);
    float*  z1     = (float*) carve((size_t)G_BATCH * FC_DIM * 4);
    float*  z2     = (float*) carve((size_t)G_BATCH * FC_DIM * 4);

    const int NF = N_NODES * F_DIM;                    // 12.8M
    const int TB = 256;
    auto blocks = [](int n, int tb) { return (n + tb - 1) / tb; };

    // 1. degrees (self-loop included) -> rsqrt
    fill_f32<<<blocks(N_NODES, TB), TB, 0, stream>>>(deg, 1.0f, N_NODES);
    deg_accum<<<blocks(N_EDGES, TB), TB, 0, stream>>>(dst, deg, N_EDGES);
    deg_rsqrt<<<blocks(N_NODES, TB), TB, 0, stream>>>(deg, N_NODES);

    // 2. bf16 feature copy of x
    cvt_f32_bf16<<<blocks(NF, TB), TB, 0, stream>>>(x, featbf, NF);

    const int gemmGrid = (N_NODES + 127) / 128;        // 782 blocks
    const int edgeGrid = 8192;

    // 3. two GCN layers
    const float* Ws[2] = { W1, W2 };
    const float* bs[2] = { b1, b2 };
    for (int layer = 0; layer < 2; ++layer) {
        pack_w_bf16<<<blocks(8 * 4 * 32 * 16, TB), TB, 0, stream>>>(Ws[layer], wpk);
        gemm_bf16_wmma<<<gemmGrid, 256, 0, stream>>>(featbf, wpk, hbuf, N_NODES);
        fill_f32<<<blocks(NF, TB), TB, 0, stream>>>(aggbuf, 0.0f, NF);
        edge_agg<<<edgeGrid, TB, 0, stream>>>((const float4*)hbuf, src, dst, deg, aggbuf);
        gcn_post<<<blocks(NF, TB), TB, 0, stream>>>(aggbuf, hbuf, deg, bs[layer], featbf);
    }

    // 4. global mean pool
    fill_f32<<<blocks(G_BATCH * F_DIM, TB), TB, 0, stream>>>(pooled, 0.0f, G_BATCH * F_DIM);
    fill_f32<<<1, TB, 0, stream>>>(cnt, 0.0f, G_BATCH);
    pool_sum<<<(N_NODES + POOL_CHUNK - 1) / POOL_CHUNK, F_DIM, 0, stream>>>(aggbuf, batch, pooled);
    pool_cnt<<<blocks(N_NODES, TB), TB, 0, stream>>>(batch, cnt);
    pool_div<<<blocks(G_BATCH * F_DIM, TB), TB, 0, stream>>>(pooled, cnt);

    // 5. MLP head
    mlp1<<<G_BATCH, FC_DIM, 0, stream>>>(pooled, sv, act, Wf1, bf1, z1);
    mlp2<<<G_BATCH, FC_DIM, 0, stream>>>(z1, Wf2, bf2, z2);
    mlp_out<<<1, G_BATCH, 0, stream>>>(z2, Wo, bo, out);
}
